// BFPDense_919123001650
// MI455X (gfx1250) — matrix operand order repaired and checked by reading, not compiled, not run on hardware
//
#include <hip/hip_runtime.h>
#include <hip/hip_bf16.h>

// BFP8 dense: quantize x and W to block-FP (8-bit mantissa, shared exp per 32).
// Dequantized values have <=8 significant bits -> exactly representable in FP16,
// so the GEMM runs on v_wmma_f32_16x16x32_f16 (f32 accumulate) with no fidelity
// loss. M=8192 (B*S), K=2048 (D), N=8192 (F). Compute-bound (~716 FLOP/B).

typedef __attribute__((ext_vector_type(16))) _Float16 v16h;
typedef __attribute__((ext_vector_type(8)))  _Float16 v8h;
typedef __attribute__((ext_vector_type(8)))  float    v8f;
typedef __attribute__((ext_vector_type(4)))  int      v4i;

#define M_DIM 8192
#define N_DIM 8192
#define K_DIM 2048

// gfx1250 async global->LDS (ASYNCcnt) support, probe-guarded
#if defined(__has_builtin)
#if __has_builtin(__builtin_amdgcn_global_load_async_to_lds_b128)
#define HAVE_ASYNC_LDS 1
#endif
#if __has_builtin(__builtin_amdgcn_s_wait_asynccnt)
#define HAVE_WAIT_ASYNC_BUILTIN 1
#endif
#endif

__device__ __forceinline__ void wait_asynccnt_le(int n) {
#if defined(HAVE_WAIT_ASYNC_BUILTIN)
  if (n == 0)  __builtin_amdgcn_s_wait_asynccnt(0);
  else         __builtin_amdgcn_s_wait_asynccnt(12);
#else
  if (n == 0)  asm volatile("s_wait_asynccnt 0x0" ::: "memory");
  else         asm volatile("s_wait_asynccnt 0xc" ::: "memory");
#endif
}

// ---------------------------------------------------------------------------
// Quantize x: blocks of 32 consecutive elements (K-contiguous), store f16 [M][K]
// ---------------------------------------------------------------------------
__global__ void bfp_quant_x(const float* __restrict__ x,
                            _Float16* __restrict__ xq, int n) {
  int i = blockIdx.x * blockDim.x + threadIdx.x;
  if (i >= n) return;
  float v = x[i];
  float m = fabsf(v);
  #pragma unroll
  for (int o = 16; o > 0; o >>= 1) m = fmaxf(m, __shfl_xor(m, o, 32));
  float safe = fmaxf(m, 1.17549435e-38f);
  int e;
  (void)frexpf(safe, &e);                  // safe = f * 2^e, f in [0.5,1)
  float scale = exp2f((float)(e - 1 - 7)); // floor(log2(safe)) - (MANT-1)
  float q = rintf(v / scale);
  q = fminf(fmaxf(q, -128.0f), 127.0f);
  float deq = (m > 0.0f) ? q * scale : 0.0f;
  xq[i] = (_Float16)deq;
}

// ---------------------------------------------------------------------------
// Quantize W [D][F] (blocks along F) and store TRANSPOSED as Wt [F][D] f16
// so GEMM B-fragments are K-contiguous per lane.
// ---------------------------------------------------------------------------
__global__ void bfp_quant_w_t(const float* __restrict__ w,
                              _Float16* __restrict__ wt) {
  __shared__ _Float16 tile[32][33];
  const int f0 = blockIdx.x * 32;
  const int d0 = blockIdx.y * 32;
  const int tx = threadIdx.x;   // f offset, lane id
  const int ty = threadIdx.y;   // d offset, wave id
  float v = w[(size_t)(d0 + ty) * N_DIM + (f0 + tx)];
  float m = fabsf(v);
  #pragma unroll
  for (int o = 16; o > 0; o >>= 1) m = fmaxf(m, __shfl_xor(m, o, 32));
  float safe = fmaxf(m, 1.17549435e-38f);
  int e;
  (void)frexpf(safe, &e);
  float scale = exp2f((float)(e - 1 - 7));
  float q = rintf(v / scale);
  q = fminf(fmaxf(q, -128.0f), 127.0f);
  float deq = (m > 0.0f) ? q * scale : 0.0f;
  tile[ty][tx] = (_Float16)deq;
  __syncthreads();
  wt[(size_t)(f0 + ty) * K_DIM + (d0 + tx)] = tile[tx][ty];
}

// ---------------------------------------------------------------------------
// GEMM: out[M][N] = Xq[M][K] * Wt[N][K]^T + bias
// 256 threads = 8 waves; block tile 128(M) x 256(N); K-step 64.
// Double-buffered LDS (96 KB) fed by GLOBAL_LOAD_ASYNC_TO_LDS_B128:
// prefetch step i+1 while WMMAs consume step i; s_wait_asynccnt <= 12 (the
// 12 chunks just issued) guarantees the previous tile's 12 chunks landed.
// ---------------------------------------------------------------------------
__global__ __launch_bounds__(256) void bfp_gemm(const _Float16* __restrict__ Aq,
                                                const _Float16* __restrict__ Bt,
                                                const float* __restrict__ bias,
                                                float* __restrict__ out) {
  __shared__ __align__(16) _Float16 As[2][128 * 64];  // 2 x 16 KB [M][K]
  __shared__ __align__(16) _Float16 Bs[2][256 * 64];  // 2 x 32 KB [N][K]

  const int tid   = threadIdx.x;
  const int lane  = tid & 31;
  const int wave  = tid >> 5;   // 0..7
  const int waveM = wave >> 2;  // 0..1 -> 64-row slice
  const int waveN = wave & 3;   // 0..3 -> 64-col slice
  const int l15   = lane & 15;
  const int lhi   = lane >> 4;  // 0 or 1

  const int mBlock = blockIdx.y * 128;
  const int nBlock = blockIdx.x * 256;

  // per-thread chunk coordinates (12 x b128 per thread per K-step)
  const int ca_r   = tid >> 3;        // A: rows 0..31 (+32 per pass), 4 passes
  const int c_col  = (tid & 7) * 8;   // 8 halfs per chunk

  v8f acc[4][4] = {};

  const int NSTEPS = K_DIM / 64;

  // ---- prefetch / load of one K-step into buffer `buf` ----
  auto load_tiles = [&](int buf, int k0) {
#if defined(HAVE_ASYNC_LDS)
    #pragma unroll
    for (int p = 0; p < 4; ++p) {
      int r = ca_r + p * 32;
      const _Float16* g = Aq + (size_t)(mBlock + r) * K_DIM + k0 + c_col;
      _Float16* l = &As[buf][r * 64 + c_col];
      __builtin_amdgcn_global_load_async_to_lds_b128((v4i*)g, (v4i*)l, 0, 0);
    }
    #pragma unroll
    for (int p = 0; p < 8; ++p) {
      int r = ca_r + p * 32;
      const _Float16* g = Bt + (size_t)(nBlock + r) * K_DIM + k0 + c_col;
      _Float16* l = &Bs[buf][r * 64 + c_col];
      __builtin_amdgcn_global_load_async_to_lds_b128((v4i*)g, (v4i*)l, 0, 0);
    }
#else
    #pragma unroll
    for (int p = 0; p < 4; ++p) {
      int r = ca_r + p * 32;
      *(uint4*)&As[buf][r * 64 + c_col] =
          *(const uint4*)(Aq + (size_t)(mBlock + r) * K_DIM + k0 + c_col);
    }
    #pragma unroll
    for (int p = 0; p < 8; ++p) {
      int r = ca_r + p * 32;
      *(uint4*)&Bs[buf][r * 64 + c_col] =
          *(const uint4*)(Bt + (size_t)(nBlock + r) * K_DIM + k0 + c_col);
    }
#endif
  };

  load_tiles(0, 0);

  for (int s = 0; s < NSTEPS; ++s) {
    const int cur = s & 1;
    if (s + 1 < NSTEPS) {
      load_tiles(cur ^ 1, (s + 1) * 64);
#if defined(HAVE_ASYNC_LDS)
      wait_asynccnt_le(12);   // previous tile's 12 chunks done; next 12 in flight
#endif
    } else {
#if defined(HAVE_ASYNC_LDS)
      wait_asynccnt_le(0);
#endif
    }
    __syncthreads();          // all waves' writes for `cur` visible

    #pragma unroll
    for (int kk = 0; kk < 64; kk += 32) {
      v16h a[4], b[4];
      // A frag (16-bit 16x32): lane<16 K{0..7,16..23}; lane>=16 K{8..15,24..31}
      #pragma unroll
      for (int mt = 0; mt < 4; ++mt) {
        const _Float16* ap =
            &As[cur][(waveM * 64 + mt * 16 + l15) * 64 + kk + lhi * 8];
        ((v8h*)&a[mt])[0] = *(const v8h*)(ap);
        ((v8h*)&a[mt])[1] = *(const v8h*)(ap + 16);
      }
      // B frag: lane<16 col=lane K{0..7,16..23}; lane>=16 col=lane-16 K{8..15,24..31}
      #pragma unroll
      for (int nt = 0; nt < 4; ++nt) {
        const _Float16* bp =
            &Bs[cur][(waveN * 64 + nt * 16 + l15) * 64 + kk + lhi * 8];
        ((v8h*)&b[nt])[0] = *(const v8h*)(bp);
        ((v8h*)&b[nt])[1] = *(const v8h*)(bp + 16);
      }
      #pragma unroll
      for (int mt = 0; mt < 4; ++mt)
        #pragma unroll
        for (int nt = 0; nt < 4; ++nt)
          acc[mt][nt] = __builtin_amdgcn_wmma_f32_16x16x32_f16(
              false, a[mt], false, b[nt], (short)0, acc[mt][nt], false, false);
    }

    __syncthreads();          // all reads of `cur` done before it is re-filled
  }

  // Epilogue: C/D layout -> VGPR r holds M = 16*mt + 8*lhi + r, N = 16*nt + l15
  #pragma unroll
  for (int nt = 0; nt < 4; ++nt) {
    int n = nBlock + waveN * 64 + nt * 16 + l15;
    float bv = bias[n];
    #pragma unroll
    for (int mt = 0; mt < 4; ++mt) {
      int mBase = mBlock + waveM * 64 + mt * 16 + lhi * 8;
      #pragma unroll
      for (int r = 0; r < 8; ++r) {
        out[(size_t)(mBase + r) * N_DIM + n] = acc[mt][nt][r] + bv;
      }
    }
  }
}

// ---------------------------------------------------------------------------
extern "C" void kernel_launch(void* const* d_in, const int* in_sizes, int n_in,
                              void* d_out, int out_size, void* d_ws,
                              size_t ws_size, hipStream_t stream) {
  (void)in_sizes; (void)n_in; (void)out_size; (void)ws_size;
  const float* x    = (const float*)d_in[0];
  const float* w    = (const float*)d_in[1];
  const float* bias = (const float*)d_in[2];
  float* out = (float*)d_out;

  _Float16* xq = (_Float16*)d_ws;                // 32 MB
  _Float16* wt = xq + (size_t)M_DIM * K_DIM;     // next 32 MB

  const int nx = M_DIM * K_DIM;
  bfp_quant_x<<<nx / 256, 256, 0, stream>>>(x, xq, nx);

  dim3 bw(32, 32);
  dim3 gw(N_DIM / 32, K_DIM / 32);
  bfp_quant_w_t<<<gw, bw, 0, stream>>>(w, wt);

  dim3 gg(N_DIM / 256, M_DIM / 128);
  bfp_gemm<<<gg, 256, 0, stream>>>(xq, wt, bias, out);
}
